// LightweightGAT_16698832847057
// MI455X (gfx1250) — compile-verified, matrix operand order checked by reading, not run on hardware
//
#include <hip/hip_runtime.h>
#include <hip/hip_bf16.h>
#include <math.h>

#define NN      50000
#define EE      800000
#define NFEAT   128
#define NHID    64
#define NCLASS  40
#define NCP     48      // NCLASS padded to multiple of 16
#define LN_EPS  1e-5f

typedef __attribute__((ext_vector_type(16))) _Float16 v16h;
typedef __attribute__((ext_vector_type(8)))  float    v8f;

// ---------------- casting / packing kernels ----------------

__global__ void k_cast_f16(const float* __restrict__ src, _Float16* __restrict__ dst, int n) {
    int i = blockIdx.x * blockDim.x + threadIdx.x;
    if (i < n) dst[i] = (_Float16)src[i];
}

// W2 (64 x 40) -> f16 padded (64 x 48)
__global__ void k_pack_w2(const float* __restrict__ W2, _Float16* __restrict__ dst) {
    int i = blockIdx.x * blockDim.x + threadIdx.x;
    if (i < NHID * NCP) {
        int k = i / NCP, n = i % NCP;
        dst[i] = (n < NCLASS) ? (_Float16)W2[k * NCLASS + n] : (_Float16)0.0f;
    }
}

// res_w (40 x 128) -> res_w^T f16 padded (128 x 48)
__global__ void k_pack_rw(const float* __restrict__ RW, _Float16* __restrict__ dst) {
    int i = blockIdx.x * blockDim.x + threadIdx.x;
    if (i < NFEAT * NCP) {
        int k = i / NCP, n = i % NCP;
        dst[i] = (n < NCLASS) ? (_Float16)RW[n * NFEAT + k] : (_Float16)0.0f;
    }
}

// relu(f32) -> f16
__global__ void k_relu_cast(const float* __restrict__ src, _Float16* __restrict__ dst, int n) {
    int i = blockIdx.x * blockDim.x + threadIdx.x;
    if (i < n) dst[i] = (_Float16)fmaxf(src[i], 0.0f);
}

// ---------------- WMMA GEMM ----------------
// C[M x NCOLS] = A[M x K](f16) * B[K x LDB](f16, zero-padded cols) (+ bias)
// One wave per (MT x 16) x 16 strip of C: B fragments are loaded once into
// registers and reused across MT M-tiles. All strides/extents compile-time.
// ISA 16-bit fragment layout (§7.12.2): lane L -> r=L%16, hi=L/16;
//   element i of v16h (v=i/2, h=i%2) holds K = hi*8 + 2*(v%4) + h + 16*(v/4).
// A: row M=r striped over K.  B mirrors A with N in place of M.
// C: VGPR j -> M = hi*8 + j, N = r.
template<int K, int LDA, int LDB, int LDC, int NCOLS, bool HASBIAS, int MT>
__global__ void k_gemm_wmma(const _Float16* __restrict__ A, const _Float16* __restrict__ B,
                            float* __restrict__ C, const float* __restrict__ bias) {
    constexpr int KT = K / 32;
    const int lane = threadIdx.x & 31;
    const int r  = lane & 15;
    const int hi = lane >> 4;
    const int n0 = blockIdx.y * 16;
    const int n  = n0 + r;

    // load B fragments for this N-tile once (reused for MT M-tiles)
    v16h bfrag[KT];
#pragma unroll
    for (int kt = 0; kt < KT; ++kt) {
#pragma unroll
        for (int i = 0; i < 16; ++i) {
            const int v = i >> 1, h = i & 1;
            const int k = hi * 8 + ((v & 3) << 1) + h + ((v >> 2) << 4);
            bfrag[kt][i] = B[(kt * 32 + k) * LDB + n];
        }
    }
    float bv = 0.0f;
    if (HASBIAS && n < NCOLS) bv = bias[n];

    const int m0base = blockIdx.x * (MT * 16);
#pragma unroll
    for (int mt = 0; mt < MT; ++mt) {
        const int m0 = m0base + mt * 16;
        v8f c = {};
#pragma unroll
        for (int kt = 0; kt < KT; ++kt) {
            v16h a;
#pragma unroll
            for (int i = 0; i < 16; ++i) {
                const int v = i >> 1, h = i & 1;
                const int k = hi * 8 + ((v & 3) << 1) + h + ((v >> 2) << 4);
                a[i] = A[(m0 + r) * LDA + kt * 32 + k];
            }
            c = __builtin_amdgcn_wmma_f32_16x16x32_f16(false, a, false, bfrag[kt],
                                                       (short)0, c, false, false);
        }
        if (n < NCOLS) {
#pragma unroll
            for (int j = 0; j < 8; ++j)
                C[(m0 + hi * 8 + j) * LDC + n] = c[j] + bv;
        }
    }
}

// ---------------- per-node attention dot products ----------------
__global__ void k_node_scores(const float* __restrict__ h1, const float* __restrict__ a1,
                              float* __restrict__ ssrc, float* __restrict__ sdst) {
    int i = blockIdx.x * blockDim.x + threadIdx.x;
    if (i < NN) {
        float s0 = 0.0f, s1 = 0.0f;
#pragma unroll 8
        for (int j = 0; j < NHID; ++j) {
            float v = h1[i * NHID + j];
            s0 += v * a1[j];
            s1 += v * a1[NHID + j];
        }
        ssrc[i] = s0;
        sdst[i] = s1;
    }
}

// ---------------- per-edge attention weight ----------------
__global__ void k_edge_w(const int* __restrict__ row, const int* __restrict__ col,
                         const float* __restrict__ adj, const float* __restrict__ ssrc,
                         const float* __restrict__ sdst, float* __restrict__ wbuf) {
    int e = blockIdx.x * blockDim.x + threadIdx.x;
    if (e < EE) {
        float sc = ssrc[row[e]] + sdst[col[e]];
        sc = (sc > 0.0f) ? sc : 0.2f * sc;              // leaky_relu(0.2)
        float sig = 1.0f / (1.0f + expf(-sc));          // sigmoid
        wbuf[e] = sig * adj[e];
    }
}

// ---------------- scatter-add layer 1: acc[row] += w * h1[col]  (64 feats, 4 threads/edge) ----------------
__global__ void k_scatter64(const int* __restrict__ row, const int* __restrict__ col,
                            const float* __restrict__ wbuf, const float* __restrict__ src,
                            float* __restrict__ acc) {
    int t = blockIdx.x * blockDim.x + threadIdx.x;
    if (t < EE * 4) {
        int e = t >> 2, q = t & 3;
        int r = row[e], c = col[e];
        float w = wbuf[e];
        const float4* s = (const float4*)(src + c * NHID + q * 16);
        float*        d = acc + r * NHID + q * 16;
#pragma unroll
        for (int v = 0; v < 4; ++v) {
            float4 sv = s[v];
            atomicAdd(d + v * 4 + 0, w * sv.x);
            atomicAdd(d + v * 4 + 1, w * sv.y);
            atomicAdd(d + v * 4 + 2, w * sv.z);
            atomicAdd(d + v * 4 + 3, w * sv.w);
        }
    }
}

// ---------------- scatter-add layer 2: acc[row] += w * h2[col]  (40 feats, 2 threads/edge) ----------------
__global__ void k_scatter40(const int* __restrict__ row, const int* __restrict__ col,
                            const float* __restrict__ wbuf, const float* __restrict__ src,
                            float* __restrict__ acc) {
    int t = blockIdx.x * blockDim.x + threadIdx.x;
    if (t < EE * 2) {
        int e = t >> 1, q = t & 1;
        int r = row[e], c = col[e];
        float w = wbuf[e];
        const float* s = src + c * NCLASS + q * 20;
        float*       d = acc + r * NCLASS + q * 20;
#pragma unroll
        for (int j = 0; j < 20; ++j) atomicAdd(d + j, w * s[j]);
    }
}

// ---------------- finalize: y = relu(h2_acc) + res; LayerNorm over 40 classes ----------------
__global__ void k_finalize(const float* __restrict__ h2acc, const float* __restrict__ res,
                           const float* __restrict__ ln_g, const float* __restrict__ ln_b,
                           float* __restrict__ out) {
    int i = blockIdx.x * blockDim.x + threadIdx.x;
    if (i < NN) {
        float y[NCLASS];
        float mu = 0.0f;
#pragma unroll
        for (int j = 0; j < NCLASS; ++j) {
            float v = fmaxf(h2acc[i * NCLASS + j], 0.0f) + res[i * NCLASS + j];
            y[j] = v;
            mu += v;
        }
        mu *= (1.0f / NCLASS);
        float var = 0.0f;
#pragma unroll
        for (int j = 0; j < NCLASS; ++j) {
            float d = y[j] - mu;
            var += d * d;
        }
        var *= (1.0f / NCLASS);
        float inv = rsqrtf(var + LN_EPS);
#pragma unroll
        for (int j = 0; j < NCLASS; ++j)
            out[i * NCLASS + j] = (y[j] - mu) * inv * ln_g[j] + ln_b[j];
    }
}

// ---------------- host launcher ----------------

extern "C" void kernel_launch(void* const* d_in, const int* in_sizes, int n_in,
                              void* d_out, int out_size, void* d_ws, size_t ws_size,
                              hipStream_t stream) {
    const float* x     = (const float*)d_in[0];
    const int*   eidx  = (const int*)  d_in[1];   // (2, E): row then col
    const float* adj   = (const float*)d_in[2];
    const float* W1    = (const float*)d_in[3];
    const float* a1    = (const float*)d_in[4];
    const float* W2    = (const float*)d_in[5];
    const float* res_w = (const float*)d_in[6];
    const float* res_b = (const float*)d_in[7];
    const float* ln_g  = (const float*)d_in[8];
    const float* ln_b  = (const float*)d_in[9];
    float* out = (float*)d_out;

    const int* row = eidx;
    const int* col = eidx + EE;

    // workspace carve-out (256B aligned)
    char* ws = (char*)d_ws;
    size_t off = 0;
    auto carve = [&](size_t bytes) -> char* {
        off = (off + 255) & ~(size_t)255;
        char* p = ws + off;
        off += bytes;
        return p;
    };
    _Float16* xh    = (_Float16*)carve((size_t)NN * NFEAT * 2);   // x in f16
    _Float16* w1h   = (_Float16*)carve((size_t)NFEAT * NHID * 2); // W1 f16
    _Float16* w2h   = (_Float16*)carve((size_t)NHID * NCP * 2);   // W2 f16 padded
    _Float16* rwh   = (_Float16*)carve((size_t)NFEAT * NCP * 2);  // res_w^T f16 padded
    float*    h1    = (float*)   carve((size_t)NN * NHID * 4);
    float*    ssrc  = (float*)   carve((size_t)NN * 4);
    float*    sdst  = (float*)   carve((size_t)NN * 4);
    float*    wbuf  = (float*)   carve((size_t)EE * 4);
    float*    h1acc = (float*)   carve((size_t)NN * NHID * 4);
    _Float16* h1rh  = (_Float16*)carve((size_t)NN * NHID * 2);    // relu(h1acc) f16
    float*    h2    = (float*)   carve((size_t)NN * NCLASS * 4);
    float*    h2acc = (float*)   carve((size_t)NN * NCLASS * 4);
    float*    resid = (float*)   carve((size_t)NN * NCLASS * 4);
    (void)ws_size;

    const int B = 256;
    constexpr int MT = 5;                 // 3125 M-tiles = 625 * 5

    // zero the atomic accumulators (every call: deterministic under graph replay)
    hipMemsetAsync(h1acc, 0, (size_t)NN * NHID * 4, stream);
    hipMemsetAsync(h2acc, 0, (size_t)NN * NCLASS * 4, stream);

    // pack inputs to f16
    k_cast_f16<<<(NN * NFEAT + B - 1) / B, B, 0, stream>>>(x, xh, NN * NFEAT);
    k_cast_f16<<<(NFEAT * NHID + B - 1) / B, B, 0, stream>>>(W1, w1h, NFEAT * NHID);
    k_pack_w2<<<(NHID * NCP + B - 1) / B, B, 0, stream>>>(W2, w2h);
    k_pack_rw<<<(NFEAT * NCP + B - 1) / B, B, 0, stream>>>(res_w, rwh);

    // h1 = x @ W1   (50000x128 * 128x64)
    k_gemm_wmma<NFEAT, NFEAT, NHID, NHID, NHID, false, MT>
        <<<dim3(NN / 16 / MT, NHID / 16), 32, 0, stream>>>(xh, w1h, h1, nullptr);

    // per-node attention dots, per-edge weights
    k_node_scores<<<(NN + B - 1) / B, B, 0, stream>>>(h1, a1, ssrc, sdst);
    k_edge_w<<<(EE + B - 1) / B, B, 0, stream>>>(row, col, adj, ssrc, sdst, wbuf);

    // layer-1 aggregation
    k_scatter64<<<(EE * 4 + B - 1) / B, B, 0, stream>>>(row, col, wbuf, h1, h1acc);

    // relu + cast, then h2 = relu(h1_out) @ W2   (K=64, N padded to 48)
    k_relu_cast<<<(NN * NHID + B - 1) / B, B, 0, stream>>>(h1acc, h1rh, NN * NHID);
    k_gemm_wmma<NHID, NHID, NCP, NCLASS, NCLASS, false, MT>
        <<<dim3(NN / 16 / MT, NCP / 16), 32, 0, stream>>>(h1rh, w2h, h2, nullptr);

    // layer-2 aggregation
    k_scatter40<<<(EE * 2 + B - 1) / B, B, 0, stream>>>(row, col, wbuf, h2, h2acc);

    // residual: resid = x @ res_w^T + res_b   (K=128, N padded to 48)
    k_gemm_wmma<NFEAT, NFEAT, NCP, NCLASS, NCLASS, true, MT>
        <<<dim3(NN / 16 / MT, NCP / 16), 32, 0, stream>>>(xh, rwh, resid, res_b);

    // relu + residual + layernorm
    k_finalize<<<(NN + B - 1) / B, B, 0, stream>>>(h2acc, resid, ln_g, ln_b, out);
}
